// HierarchicalPredictiveCoding_87308095193493
// MI455X (gfx1250) — matrix operand order repaired and checked by reading, not compiled, not run on hardware
//
#include <hip/hip_runtime.h>
#include <hip/hip_bf16.h>
#include <math.h>

#define THREADS 256

typedef __attribute__((ext_vector_type(16))) __bf16 v16bf;
typedef __attribute__((ext_vector_type(8)))  float  v8f;

__device__ __forceinline__ unsigned short f2bf(float f) {
  union { float f; unsigned u; } c; c.f = f;
  unsigned u = c.u;
  return (unsigned short)((u + 0x7FFFu + ((u >> 16) & 1u)) >> 16);
}
__device__ __forceinline__ float bf2f(unsigned short h) {
  union { unsigned u; float f; } c; c.u = ((unsigned)h) << 16;
  return c.f;
}

// CDNA5 async global->LDS copy (16B per lane, tracked by ASYNCcnt), and its wait.
// Inline asm so it works on both ROCm-7.2 and amdgpu-toolchain builtin arities.
__device__ __forceinline__ void async_copy_b128(unsigned ldsOff, const unsigned short* g) {
  asm volatile("global_load_async_to_lds_b128 %0, %1, off"
               :: "v"(ldsOff), "v"((unsigned long long)(size_t)g)
               : "memory");
}
__device__ __forceinline__ void wait_async_zero() {
  asm volatile("s_wait_asynccnt 0" ::: "memory");
}

// ---------------------------------------------------------------------------
// WMMA GEMM: C[M,N] = op(A)[M,K] * op(B)[K,N]  (bf16 in, f32 accumulate)
//   TA==0: op(A)[m,k] = A[m*lda+k]      TA==1: op(A)[m,k] = A[k*lda+m]
//   TB==0: op(B)[k,n] = B[k*ldb+n]      TB==1: op(B)[k,n] = B[n*ldb+k]
// Double-buffered LDS staging; contiguous operands staged with
// GLOBAL_LOAD_ASYNC_TO_LDS_B128 + s_wait_asynccnt, transposed operands staged
// with a synchronous scalar transpose overlapped with WMMA on the other buffer.
// Epilogues (template EPI):
//   0: outF = C + bias[n]                           (REC, f32)
//   1: outH = bf16(tanh(C + bias[n]))               (H)
//   2: outH = bf16(C * (1 - h^2)), h = hbuf[m,n]    (DA)
//   3: wMaster -= alpha*C; wShadow = bf16(wMaster)  (fused SGD on rW1/rW2)
// M,N multiples of 128; K multiple of 32.
// ---------------------------------------------------------------------------
template<int EPI, int TA, int TB>
__global__ __launch_bounds__(THREADS)
void gemm_wmma(const unsigned short* __restrict__ A, int lda,
               const unsigned short* __restrict__ B, int ldb,
               int M, int N, int K,
               const float* __restrict__ bias,
               const unsigned short* __restrict__ hbuf,
               float* __restrict__ outF,
               unsigned short* __restrict__ outH,
               float* __restrict__ wMaster,
               unsigned short* __restrict__ wShadow,
               float alpha)
{
  __shared__ __align__(16) unsigned short At[2][128][40];
  __shared__ __align__(16) unsigned short Bt[2][128][40];
  const int tid  = threadIdx.x;
  const int wave = tid >> 5, lane = tid & 31;
  const int half = lane >> 4, lr = lane & 15;
  const int wm = wave >> 1, wn = wave & 1;      // 4x2 wave grid
  const int m0w = wm * 32, n0w = wn * 64;       // 32x64 per wave
  const int rowBase = blockIdx.y * 128;
  const int colBase = blockIdx.x * 128;

  // Stage tile k0 into buffer buf (512 16B slots per operand, 2 per thread).
  auto stage = [&](int k0, int buf) {
    if (TA == 0) {
#pragma unroll
      for (int rep = 0; rep < 2; ++rep) {
        int s = tid + rep * THREADS;
        int m = s >> 2, kq = (s & 3) << 3;
        async_copy_b128((unsigned)(size_t)&At[buf][m][kq],
                        A + (size_t)(rowBase + m) * lda + k0 + kq);
      }
    } else {
#pragma unroll
      for (int rep = 0; rep < 2; ++rep) {
        int s = tid + rep * THREADS;
        int k = s >> 4, mq = (s & 15) << 3;
        uint4 v = *reinterpret_cast<const uint4*>(A + (size_t)(k0 + k) * lda + rowBase + mq);
        const unsigned short* e = reinterpret_cast<const unsigned short*>(&v);
#pragma unroll
        for (int i = 0; i < 8; ++i) At[buf][mq + i][k] = e[i];
      }
    }
    if (TB == 1) {
#pragma unroll
      for (int rep = 0; rep < 2; ++rep) {
        int s = tid + rep * THREADS;
        int n = s >> 2, kq = (s & 3) << 3;
        async_copy_b128((unsigned)(size_t)&Bt[buf][n][kq],
                        B + (size_t)(colBase + n) * ldb + k0 + kq);
      }
    } else {
#pragma unroll
      for (int rep = 0; rep < 2; ++rep) {
        int s = tid + rep * THREADS;
        int k = s >> 4, nq = (s & 15) << 3;
        uint4 v = *reinterpret_cast<const uint4*>(B + (size_t)(k0 + k) * ldb + colBase + nq);
        const unsigned short* e = reinterpret_cast<const unsigned short*>(&v);
#pragma unroll
        for (int i = 0; i < 8; ++i) Bt[buf][nq + i][k] = e[i];
      }
    }
  };

  v8f acc[2][4];
#pragma unroll
  for (int mi = 0; mi < 2; ++mi)
#pragma unroll
    for (int ni = 0; ni < 4; ++ni)
#pragma unroll
      for (int e = 0; e < 8; ++e) acc[mi][ni][e] = 0.0f;

  stage(0, 0);
  const int nk = K >> 5;
  for (int i = 0; i < nk; ++i) {
    const int buf = i & 1;
    if (TA == 0 || TB == 1) wait_async_zero();  // this wave's tile-i copies landed
    __syncthreads();                            // all waves' tile i visible; tile i-1 reads done
    if (i + 1 < nk) stage((i + 1) << 5, buf ^ 1);

    // ---- fragments per ISA 7.12.2 (16-bit A 16x32; B staged as [n][k])
    union FC { uint4 q[2]; v16bf v; };
    FC a[2], b[4];
#pragma unroll
    for (int mi = 0; mi < 2; ++mi) {
      int m = m0w + 16 * mi + lr;
      int kb = 8 * half; // lanes0-15: K 0..7 & 16..23 ; lanes16-31: K 8..15 & 24..31
      a[mi].q[0] = *reinterpret_cast<const uint4*>(&At[buf][m][kb]);
      a[mi].q[1] = *reinterpret_cast<const uint4*>(&At[buf][m][kb + 16]);
    }
#pragma unroll
    for (int ni = 0; ni < 4; ++ni) {
      int n = n0w + 16 * ni + lr;
      int kb = 8 * half; // lanes0-15: K 0..7 & 16..23 ; lanes16-31: K 8..15 & 24..31
      b[ni].q[0] = *reinterpret_cast<const uint4*>(&Bt[buf][n][kb]);
      b[ni].q[1] = *reinterpret_cast<const uint4*>(&Bt[buf][n][kb + 16]);
    }
#pragma unroll
    for (int mi = 0; mi < 2; ++mi)
#pragma unroll
      for (int ni = 0; ni < 4; ++ni)
        acc[mi][ni] = __builtin_amdgcn_wmma_f32_16x16x32_bf16(
            false, a[mi].v, false, b[ni].v, (short)0, acc[mi][ni], false, false);
  }

  // ---- epilogue (C layout: VGPR r -> M = r + 8*half, N = lane%16)
#pragma unroll
  for (int mi = 0; mi < 2; ++mi)
#pragma unroll
    for (int ni = 0; ni < 4; ++ni)
#pragma unroll
      for (int r = 0; r < 8; ++r) {
        int gm = rowBase + m0w + 16 * mi + 8 * half + r;
        int gn = colBase + n0w + 16 * ni + lr;
        size_t idx = (size_t)gm * N + gn;
        float c = acc[mi][ni][r];
        if (EPI == 0) {
          outF[idx] = c + bias[gn];
        } else if (EPI == 1) {
          outH[idx] = f2bf(tanhf(c + bias[gn]));
        } else if (EPI == 2) {
          float h = bf2f(hbuf[idx]);
          outH[idx] = f2bf(c * (1.0f - h * h));
        } else {
          float w = wMaster[idx] - alpha * c;
          wMaster[idx] = w;
          wShadow[idx] = f2bf(w);
        }
      }
}

// ---------------------------------------------------------------------------
// Small helper kernels (exact f32 path for all vector / rank-1 math)
// ---------------------------------------------------------------------------
__global__ __launch_bounds__(THREADS) void k_f2bf_arr(unsigned short* dst, const float* src, size_t n) {
  size_t i = (size_t)blockIdx.x * blockDim.x + threadIdx.x;
  size_t st = (size_t)gridDim.x * blockDim.x;
  for (; i < n; i += st) dst[i] = f2bf(src[i]);
}
__global__ __launch_bounds__(THREADS) void k_fill(float* p, float v, size_t n) {
  size_t i = (size_t)blockIdx.x * blockDim.x + threadIdx.x;
  size_t st = (size_t)gridDim.x * blockDim.x;
  for (; i < n; i += st) p[i] = v;
}
// column mean & mean-of-square of X (done once)
__global__ __launch_bounds__(THREADS) void k_colstats(const float* X, int rows, int cols,
                                                      float* xbar, float* xsq) {
  int c = blockIdx.x * blockDim.x + threadIdx.x;
  if (c >= cols) return;
  float s = 0.f, s2 = 0.f;
  for (int r = 0; r < rows; ++r) { float v = X[(size_t)r * cols + c]; s += v; s2 += v * v; }
  float inv = 1.0f / rows;
  xbar[c] = s * inv; xsq[c] = s2 * inv;
}
__global__ __launch_bounds__(THREADS) void k_colsum_f32(const float* M, int rows, int cols,
                                                        float* out, float scale) {
  int c = blockIdx.x * blockDim.x + threadIdx.x;
  if (c >= cols) return;
  float s = 0.f;
  for (int r = 0; r < rows; ++r) s += M[(size_t)r * cols + c];
  out[c] = s * scale;
}
__global__ __launch_bounds__(THREADS) void k_colsum_bf16(const unsigned short* M, int rows, int cols,
                                                         float* out, float scale) {
  int c = blockIdx.x * blockDim.x + threadIdx.x;
  if (c >= cols) return;
  float s = 0.f;
  for (int r = 0; r < rows; ++r) s += bf2f(M[(size_t)r * cols + c]);
  out[c] = s * scale;
}
// y[row] = (tanh?)(sum_k W[row,k]*x[k] + b[row]) ; one block per row
__global__ __launch_bounds__(THREADS) void k_matvec(const float* W, const float* x, const float* b,
                                                    float* y, int cols, int tanhFlag) {
  __shared__ float red[THREADS];
  int row = blockIdx.x;
  float s = 0.f;
  for (int k = threadIdx.x; k < cols; k += THREADS) s += W[(size_t)row * cols + k] * x[k];
  red[threadIdx.x] = s; __syncthreads();
  for (int o = THREADS / 2; o > 0; o >>= 1) {
    if (threadIdx.x < o) red[threadIdx.x] += red[threadIdx.x + o];
    __syncthreads();
  }
  if (threadIdx.x == 0) { float v = red[0] + b[row]; y[row] = tanhFlag ? tanhf(v) : v; }
}
// y[col] = sum_j W[j,col]*u[j]
__global__ __launch_bounds__(THREADS) void k_matvecT(const float* W, const float* u, float* y,
                                                     int rows, int cols) {
  int c = blockIdx.x * blockDim.x + threadIdx.x;
  if (c >= cols) return;
  float s = 0.f;
  for (int j = 0; j < rows; ++j) s += W[(size_t)j * cols + c] * u[j];
  y[c] = s;
}
// W[r,c] += s*u[r]*v[c]
__global__ __launch_bounds__(THREADS) void k_rank1(float* W, const float* u, const float* v,
                                                   float s, int rows, int cols) {
  size_t i = (size_t)blockIdx.x * blockDim.x + threadIdx.x;
  if (i >= (size_t)rows * cols) return;
  int r = (int)(i / cols), c = (int)(i % cols);
  W[i] += s * u[r] * v[c];
}
__global__ __launch_bounds__(THREADS) void k_axpy(float* y, const float* x, float s, int n) {
  int i = blockIdx.x * blockDim.x + threadIdx.x;
  if (i < n) y[i] += s * x[i];
}
__global__ __launch_bounds__(THREADS) void k_sub_scale(float* out, const float* a, const float* b,
                                                       float s, int n) {
  int i = blockIdx.x * blockDim.x + threadIdx.x;
  if (i < n) out[i] = s * (a[i] - b[i]);
}
__global__ __launch_bounds__(THREADS) void k_dtanh(float* d, const float* h, int n) {
  int i = blockIdx.x * blockDim.x + threadIdx.x;
  if (i < n) d[i] *= (1.0f - h[i] * h[i]);
}
// scale = clip(mean(prec_old), 0.1, 10)
__global__ __launch_bounds__(THREADS) void k_mean_clip(const float* v, int n, float* out) {
  __shared__ float red[THREADS];
  float s = 0.f;
  for (int i = threadIdx.x; i < n; i += THREADS) s += v[i];
  red[threadIdx.x] = s; __syncthreads();
  for (int o = THREADS / 2; o > 0; o >>= 1) {
    if (threadIdx.x < o) red[threadIdx.x] += red[threadIdx.x + o];
    __syncthreads();
  }
  if (threadIdx.x == 0) { float m = red[0] / n; *out = fminf(fmaxf(m, 0.1f), 10.0f); }
}
// level-0: inst_j = E[x^2]_j - 2 p_j E[x]_j + p_j^2
__global__ __launch_bounds__(THREADS) void k_inst0(float* inst, const float* xbar, const float* xsq,
                                                   const float* p, int n) {
  int i = blockIdx.x * blockDim.x + threadIdx.x;
  if (i < n) inst[i] = xsq[i] - 2.0f * p[i] * xbar[i] + p[i] * p[i];
}
__global__ __launch_bounds__(THREADS) void k_inst_vec(float* inst, const float* x, const float* p, int n) {
  int i = blockIdx.x * blockDim.x + threadIdx.x;
  if (i < n) { float d = x[i] - p[i]; inst[i] = d * d; }
}
__global__ __launch_bounds__(THREADS) void k_ema(float* ev, float* prec, const float* inst, int n) {
  int i = blockIdx.x * blockDim.x + threadIdx.x;
  if (i < n) {
    float e = ev[i] * 0.95f + 0.05f * inst[i];
    ev[i] = e;
    prec[i] = fminf(1.0f / (e + 1e-6f), 100.0f);
  }
}
// beliefs += LR * scale * (rec - beliefs)
__global__ __launch_bounds__(THREADS) void k_lerp(float* y, const float* t, const float* scale,
                                                  float lr, int n) {
  int i = blockIdx.x * blockDim.x + threadIdx.x;
  if (i < n) y[i] += lr * (*scale) * (t[i] - y[i]);
}
// Draw = bf16(REC - bn[j])
__global__ __launch_bounds__(THREADS) void k_draw(unsigned short* draw, const float* rec,
                                                  const float* bn, size_t total, int cols) {
  size_t i = (size_t)blockIdx.x * blockDim.x + threadIdx.x;
  size_t st = (size_t)gridDim.x * blockDim.x;
  for (; i < total; i += st) draw[i] = f2bf(rec[i] - bn[i % cols]);
}
__global__ __launch_bounds__(THREADS) void k_broadcast(float* out, const float* p, size_t total, int cols) {
  size_t i = (size_t)blockIdx.x * blockDim.x + threadIdx.x;
  size_t st = (size_t)gridDim.x * blockDim.x;
  for (; i < total; i += st) out[i] = p[i % cols];
}
__global__ __launch_bounds__(THREADS) void k_abs_err_big(const float* X, const float* p, int cols,
                                                         size_t total, float* out) {
  __shared__ float red[THREADS];
  size_t i = (size_t)blockIdx.x * blockDim.x + threadIdx.x;
  size_t st = (size_t)gridDim.x * blockDim.x;
  float s = 0.f;
  for (; i < total; i += st) s += fabsf(X[i] - p[i % cols]);
  red[threadIdx.x] = s; __syncthreads();
  for (int o = THREADS / 2; o > 0; o >>= 1) {
    if (threadIdx.x < o) red[threadIdx.x] += red[threadIdx.x + o];
    __syncthreads();
  }
  if (threadIdx.x == 0) atomicAdd(out, red[0] / (float)total);
}
__global__ __launch_bounds__(THREADS) void k_abs_err_vec(const float* x, const float* p, int n, float* out) {
  __shared__ float red[THREADS];
  float s = 0.f;
  for (int i = threadIdx.x; i < n; i += THREADS) s += fabsf(x[i] - p[i]);
  red[threadIdx.x] = s; __syncthreads();
  for (int o = THREADS / 2; o > 0; o >>= 1) {
    if (threadIdx.x < o) red[threadIdx.x] += red[threadIdx.x + o];
    __syncthreads();
  }
  if (threadIdx.x == 0) atomicAdd(out, red[0] / (float)n);
}

// ---------------------------------------------------------------------------
// Host side
// ---------------------------------------------------------------------------
static inline float* wsF(char* base, size_t& off, size_t n) {
  off = (off + 255) & ~(size_t)255;
  float* p = reinterpret_cast<float*>(base + off);
  off += n * sizeof(float);
  return p;
}
static inline unsigned short* wsH(char* base, size_t& off, size_t n) {
  off = (off + 255) & ~(size_t)255;
  unsigned short* p = reinterpret_cast<unsigned short*>(base + off);
  off += n * sizeof(unsigned short);
  return p;
}
static inline unsigned gdiv(int n) { return (unsigned)((n + THREADS - 1) / THREADS); }

static const float C_LR = 0.1f, C_GENLR = 0.01f;

// One step of a rank-1 (broadcast-input) level i>0. Exact f32 math.
static void rank1_step(float* Pp[8], int ind, int bd, const float* xvec,
                       float* beliefs, float* ev, float* prec, float* scale,
                       float** T, bool fin, float* errSlot, hipStream_t s)
{
  unsigned gi = gdiv(ind), gb = gdiv(bd);
  k_matvec<<<bd, THREADS, 0, s>>>(Pp[0], beliefs, Pp[1], T[0], bd, 1);      // h_g
  k_matvec<<<ind, THREADS, 0, s>>>(Pp[2], T[0], Pp[3], T[1], bd, 0);        // pred
  k_mean_clip<<<1, THREADS, 0, s>>>(prec, ind, scale);                      // scale (prec_old)
  k_inst_vec<<<gi, THREADS, 0, s>>>(T[2], xvec, T[1], ind);
  k_ema<<<gi, THREADS, 0, s>>>(ev, prec, T[2], ind);
  if (fin) k_abs_err_vec<<<1, THREADS, 0, s>>>(xvec, T[1], ind, errSlot);
  k_matvec<<<bd, THREADS, 0, s>>>(Pp[4], xvec, Pp[5], T[3], ind, 1);        // h_r
  k_matvec<<<bd, THREADS, 0, s>>>(Pp[6], T[3], Pp[7], T[4], bd, 0);         // rec (== rec_t)
  k_lerp<<<gb, THREADS, 0, s>>>(beliefs, T[4], scale, C_LR, bd);            // beliefs -> bn
  // generative grads (at bn, old params)
  k_matvec<<<bd, THREADS, 0, s>>>(Pp[0], beliefs, Pp[1], T[5], bd, 1);      // h2
  k_matvec<<<ind, THREADS, 0, s>>>(Pp[2], T[5], Pp[3], T[6], bd, 0);        // pred2
  k_sub_scale<<<gi, THREADS, 0, s>>>(T[7], T[6], xvec, 2.0f / ind, ind);    // dpred
  k_matvecT<<<gb, THREADS, 0, s>>>(Pp[2], T[7], T[8], ind, bd);             // dh (old gW2)
  k_dtanh<<<gb, THREADS, 0, s>>>(T[8], T[5], bd);                           // da
  k_rank1<<<gdiv(ind * bd), THREADS, 0, s>>>(Pp[2], T[7], T[5], -C_GENLR, ind, bd);
  k_axpy<<<gi, THREADS, 0, s>>>(Pp[3], T[7], -C_GENLR, ind);
  k_rank1<<<gdiv(bd * bd), THREADS, 0, s>>>(Pp[0], T[8], beliefs, -C_GENLR, bd, bd);
  k_axpy<<<gb, THREADS, 0, s>>>(Pp[1], T[8], -C_GENLR, bd);
  // recognition grads
  k_sub_scale<<<gb, THREADS, 0, s>>>(T[9], T[4], beliefs, 2.0f / bd, bd);   // drec
  k_matvecT<<<gb, THREADS, 0, s>>>(Pp[6], T[9], T[10], bd, bd);             // dhr (old rW2)
  k_dtanh<<<gb, THREADS, 0, s>>>(T[10], T[3], bd);
  k_rank1<<<gdiv(bd * bd), THREADS, 0, s>>>(Pp[6], T[9], T[3], -C_GENLR, bd, bd);
  k_axpy<<<gb, THREADS, 0, s>>>(Pp[7], T[9], -C_GENLR, bd);
  k_rank1<<<gdiv(bd * ind), THREADS, 0, s>>>(Pp[4], T[10], xvec, -C_GENLR, bd, ind);
  k_axpy<<<gb, THREADS, 0, s>>>(Pp[5], T[10], -C_GENLR, bd);
}

extern "C" void kernel_launch(void* const* d_in, const int* in_sizes, int n_in,
                              void* d_out, int out_size, void* d_ws, size_t ws_size,
                              hipStream_t stream)
{
  (void)out_size; (void)ws_size;
  const int B = 4096, D0 = 1024;
  const int indv[3] = {1024, 1024, 512};
  const int bdv[3]  = {1024, 512, 256};
  const float* X = (const float*)d_in[0];
  float* OUT = (float*)d_out;
  float* errSlot = OUT + (size_t)B * D0;

  // ---- workspace layout (recomputed every call; deterministic)
  char* base = (char*)d_ws; size_t off = 0;
  float* P[3][8];
  size_t psz[3][8];
  for (int L = 0; L < 3; ++L) {
    int ind = indv[L], bd = bdv[L];
    size_t sz[8] = {(size_t)bd * bd, (size_t)bd, (size_t)ind * bd, (size_t)ind,
                    (size_t)bd * ind, (size_t)bd, (size_t)bd * bd, (size_t)bd};
    for (int k = 0; k < 8; ++k) { psz[L][k] = sz[k]; P[L][k] = wsF(base, off, sz[k]); }
  }
  size_t nX = (size_t)B * D0;
  unsigned short* Xbf    = wsH(base, off, nX);
  unsigned short* Hbf    = wsH(base, off, nX);
  unsigned short* Drawbf = wsH(base, off, nX);
  unsigned short* DAbf   = wsH(base, off, nX);
  unsigned short* rW1bf  = wsH(base, off, (size_t)D0 * D0);
  unsigned short* rW2bf  = wsH(base, off, (size_t)D0 * D0);
  float* REC  = wsF(base, off, nX);
  float* xbar = wsF(base, off, D0);
  float* xsq  = wsF(base, off, D0);
  float* bel[3]  = {wsF(base, off, 1024), wsF(base, off, 512), wsF(base, off, 256)};
  float* ev[3]   = {wsF(base, off, 1024), wsF(base, off, 1024), wsF(base, off, 512)};
  float* prec[3] = {wsF(base, off, 1024), wsF(base, off, 1024), wsF(base, off, 512)};
  float* scales  = wsF(base, off, 8);
  float* pv0     = wsF(base, off, 1024);   // level-0 pred vector (survives to output)
  float* rect    = wsF(base, off, 1024);
  float* T[12];
  for (int i = 0; i < 12; ++i) T[i] = wsF(base, off, 1024);

  // ---- copy params (never mutate d_in); detect flatten order from in_sizes
  bool sortedLayout = (n_in >= 3 && in_sizes[2] > 4096); // insertion: idx2==gb1(1024); sorted: idx2==gW2
  const int perm_ins[8] = {0, 1, 2, 3, 4, 5, 6, 7};
  const int perm_srt[8] = {0, 2, 1, 3, 4, 6, 5, 7}; // canonical -> input offset
  const int* perm = sortedLayout ? perm_srt : perm_ins;
  for (int L = 0; L < 3; ++L)
    for (int k = 0; k < 8; ++k)
      hipMemcpyAsync(P[L][k], d_in[1 + 8 * L + perm[k]], psz[L][k] * sizeof(float),
                     hipMemcpyDeviceToDevice, stream);

  // ---- one-time prep
  k_f2bf_arr<<<4096, THREADS, 0, stream>>>(Xbf, X, nX);
  k_f2bf_arr<<<4096, THREADS, 0, stream>>>(rW1bf, P[0][4], (size_t)D0 * D0);
  k_f2bf_arr<<<4096, THREADS, 0, stream>>>(rW2bf, P[0][6], (size_t)D0 * D0);
  k_colstats<<<gdiv(D0), THREADS, 0, stream>>>(X, B, D0, xbar, xsq);
  k_fill<<<4, THREADS, 0, stream>>>(bel[0], 0.f, 1024);
  k_fill<<<2, THREADS, 0, stream>>>(bel[1], 0.f, 512);
  k_fill<<<1, THREADS, 0, stream>>>(bel[2], 0.f, 256);
  for (int L = 0; L < 3; ++L) {
    k_fill<<<gdiv(indv[L]), THREADS, 0, stream>>>(ev[L], 1.f, indv[L]);
    k_fill<<<gdiv(indv[L]), THREADS, 0, stream>>>(prec[L], 1.f, indv[L]);
  }
  k_fill<<<1, THREADS, 0, stream>>>(errSlot, 0.f, 1);

  const float alphaW = C_GENLR * 2.0f / ((float)B * D0);
  dim3 gBig(D0 / 128, B / 128);   // 8 x 32
  dim3 gSq(D0 / 128, D0 / 128);   // 8 x 8

  for (int it = 0; it < 3; ++it) {
    bool fin = (it == 2);
    // ===================== level 0 =====================
    k_matvec<<<D0, THREADS, 0, stream>>>(P[0][0], bel[0], P[0][1], T[0], D0, 1);  // h_g
    k_matvec<<<D0, THREADS, 0, stream>>>(P[0][2], T[0], P[0][3], pv0, D0, 0);     // pred (vector)
    k_mean_clip<<<1, THREADS, 0, stream>>>(prec[0], D0, scales + 0);
    k_inst0<<<gdiv(D0), THREADS, 0, stream>>>(T[1], xbar, xsq, pv0, D0);
    k_ema<<<gdiv(D0), THREADS, 0, stream>>>(ev[0], prec[0], T[1], D0);
    if (fin) k_abs_err_big<<<512, THREADS, 0, stream>>>(X, pv0, D0, nX, errSlot);
    // H = tanh(X rW1^T + rb1)   [WMMA, async-staged]
    gemm_wmma<1, 0, 1><<<gBig, THREADS, 0, stream>>>(Xbf, D0, rW1bf, D0, B, D0, D0,
        P[0][5], nullptr, nullptr, Hbf, nullptr, nullptr, 0.f);
    // REC = H rW2^T + rb2       [WMMA, async-staged]
    gemm_wmma<0, 0, 1><<<gBig, THREADS, 0, stream>>>(Hbf, D0, rW2bf, D0, B, D0, D0,
        P[0][7], nullptr, REC, nullptr, nullptr, nullptr, 0.f);
    k_colsum_f32<<<gdiv(D0), THREADS, 0, stream>>>(REC, B, D0, rect, 1.0f / B);   // rec_t
    k_lerp<<<gdiv(D0), THREADS, 0, stream>>>(bel[0], rect, scales + 0, C_LR, D0); // beliefs -> bn
    // generative grads (rank-1, via xbar)
    k_matvec<<<D0, THREADS, 0, stream>>>(P[0][0], bel[0], P[0][1], T[2], D0, 1);  // h2
    k_matvec<<<D0, THREADS, 0, stream>>>(P[0][2], T[2], P[0][3], T[3], D0, 0);    // pred2
    k_sub_scale<<<gdiv(D0), THREADS, 0, stream>>>(T[4], T[3], xbar, 2.0f / D0, D0);
    k_matvecT<<<gdiv(D0), THREADS, 0, stream>>>(P[0][2], T[4], T[5], D0, D0);
    k_dtanh<<<gdiv(D0), THREADS, 0, stream>>>(T[5], T[2], D0);
    k_rank1<<<gdiv(D0 * D0), THREADS, 0, stream>>>(P[0][2], T[4], T[2], -C_GENLR, D0, D0);
    k_axpy<<<gdiv(D0), THREADS, 0, stream>>>(P[0][3], T[4], -C_GENLR, D0);
    k_rank1<<<gdiv(D0 * D0), THREADS, 0, stream>>>(P[0][0], T[5], bel[0], -C_GENLR, D0, D0);
    k_axpy<<<gdiv(D0), THREADS, 0, stream>>>(P[0][1], T[5], -C_GENLR, D0);
    // recognition grads (batch, via WMMA); Draw = REC - bn
    k_draw<<<4096, THREADS, 0, stream>>>(Drawbf, REC, bel[0], nX, D0);
    // DA = (Draw rW2) * (1-H^2)  [reads OLD rW2bf -> must run before SGD on rW2]
    gemm_wmma<2, 0, 0><<<gBig, THREADS, 0, stream>>>(Drawbf, D0, rW2bf, D0, B, D0, D0,
        nullptr, Hbf, nullptr, DAbf, nullptr, nullptr, 0.f);
    // rW2 -= alpha * Draw^T H (fused SGD + bf16 shadow refresh)
    gemm_wmma<3, 1, 0><<<gSq, THREADS, 0, stream>>>(Drawbf, D0, Hbf, D0, D0, D0, B,
        nullptr, nullptr, nullptr, nullptr, P[0][6], rW2bf, alphaW);
    // rb2 -= GEN_LR * (2/bd) * (rec_t - bn)   [colsum(Draw) shortcut]
    k_sub_scale<<<gdiv(D0), THREADS, 0, stream>>>(T[6], rect, bel[0], 2.0f / D0, D0);
    k_axpy<<<gdiv(D0), THREADS, 0, stream>>>(P[0][7], T[6], -C_GENLR, D0);
    // rW1 -= alpha * DA^T X
    gemm_wmma<3, 1, 0><<<gSq, THREADS, 0, stream>>>(DAbf, D0, Xbf, D0, D0, D0, B,
        nullptr, nullptr, nullptr, nullptr, P[0][4], rW1bf, alphaW);
    // rb1 -= GEN_LR * (2/(B*bd)) * colsum(DA)
    k_colsum_bf16<<<gdiv(D0), THREADS, 0, stream>>>(DAbf, B, D0, T[7], 2.0f / ((float)B * D0));
    k_axpy<<<gdiv(D0), THREADS, 0, stream>>>(P[0][5], T[7], -C_GENLR, D0);

    // ===================== levels 1, 2 (rank-1) =====================
    rank1_step(P[1], indv[1], bdv[1], bel[0], bel[1], ev[1], prec[1], scales + 1, T, fin, errSlot, stream);
    rank1_step(P[2], indv[2], bdv[2], bel[1], bel[2], ev[2], prec[2], scales + 2, T, fin, errSlot, stream);
  }

  // ---- outputs: pred0 broadcast, total_error (accumulated), top beliefs
  k_broadcast<<<4096, THREADS, 0, stream>>>(OUT, pv0, nX, D0);
  hipMemcpyAsync(OUT + nX + 1, bel[2], 256 * sizeof(float), hipMemcpyDeviceToDevice, stream);
}
